// MHA_User_Encoder_59596966199868
// MI455X (gfx1250) — matrix operand order repaired and checked by reading, not compiled, not run on hardware
//
#include <hip/hip_runtime.h>
#include <hip/hip_bf16.h>
#include <stdint.h>
#include <stddef.h>

// Problem constants
#define B_   256
#define L_   200
#define D_   256
#define H_   16
#define DH_  16
#define LP_  208   // L padded to multiple of 16 (13 tiles)
#define LPK_ 224   // L padded to multiple of 32 (K-dim of P@V)

typedef __bf16 bf16_t;
typedef __bf16 v8bf  __attribute__((ext_vector_type(8)));
typedef __bf16 v16bf __attribute__((ext_vector_type(16)));
typedef float  v8f   __attribute__((ext_vector_type(8)));

static __device__ __forceinline__ v8f wmma_bf16(v16bf a, v16bf b, v8f c) {
  // D = A(16x32 bf16) * B(32x16 bf16) + C(16x16 f32)
  return __builtin_amdgcn_wmma_f32_16x16x32_bf16(false, a, false, b, (short)0, c,
                                                 false, false);
}

// A fragment: 16(M) x 32(K) bf16. Source rows are K-contiguous with rowStride elems.
// Per ISA 7.12.2: lanes 0-15 M=0..15 hold K {0..7,16..23}; lanes 16-31 hold K {8..15,24..31}.
static __device__ __forceinline__ v16bf load_a_frag(const bf16_t* base, int rowStride, int lane) {
  int m   = lane & 15;
  int off = (lane < 16) ? 0 : 8;
  const bf16_t* p = base + m * rowStride;
  v8bf lo = *(const v8bf*)(p + off);        // K = off .. off+7       -> VGPR0..3
  v8bf hi = *(const v8bf*)(p + 16 + off);   // K = 16+off .. 16+off+7 -> VGPR4..7
  v16bf a;
#pragma unroll
  for (int i = 0; i < 8; ++i) { a[i] = lo[i]; a[8 + i] = hi[i]; }
  return a;
}

// B fragment: 32(K) x 16(N) bf16 from K-contiguous [n][k] storage (rowStride elems per n).
// Per ISA 7.12.2: lane n = lane%16; lanes 0-15 hold K {0..7,16..23}; lanes 16-31 hold K {8..15,24..31}.
static __device__ __forceinline__ v16bf load_b_frag(const bf16_t* base, int rowStride, int lane) {
  int n  = lane & 15;
  int k0 = (lane < 16) ? 0 : 8;
  const bf16_t* p = base + n * rowStride + k0;
  v8bf lo = *(const v8bf*)(p);
  v8bf hi = *(const v8bf*)(p + 16);
  v16bf b;
#pragma unroll
  for (int i = 0; i < 8; ++i) { b[i] = lo[i]; b[8 + i] = hi[i]; }
  return b;
}

// ---------------------------------------------------------------- converts
__global__ __launch_bounds__(256) void cvt_f32_to_bf16_kernel(
    const float* __restrict__ src, bf16_t* __restrict__ dst, int n) {
  for (int i = blockIdx.x * 256 + threadIdx.x; i < n; i += gridDim.x * 256)
    dst[i] = (bf16_t)src[i];
}

// Wt[n*256+k] = W[k*256+n]  (weights stored transposed, K-contiguous, for B frags)
__global__ __launch_bounds__(256) void cvt_w_transpose_kernel(
    const float* __restrict__ W, bf16_t* __restrict__ Wt) {
  int idx = blockIdx.x * 256 + threadIdx.x;   // grid = 256 blocks -> 65536 threads
  int k = idx >> 8, n = idx & 255;
  Wt[(size_t)n * 256 + k] = (bf16_t)W[idx];
}

// ---------------------------------------------------------------- GEMM
// C[M x 256] = A[M x 256](bf16) @ W(orig 256x256, given transposed [n][k] bf16) + bias
// Block: 256 threads (8 waves), tile 64 rows x 256 cols, K-steps of 32.
__global__ __launch_bounds__(256) void gemm_bf16_kernel(
    const bf16_t* __restrict__ A, const bf16_t* __restrict__ Wt_g,
    const float* __restrict__ bias, bf16_t* __restrict__ outBf,
    float* __restrict__ outF) {
  __shared__ bf16_t As[64][32];     // row-major, K contiguous
  __shared__ bf16_t Wt[256][32];    // [n][k] K contiguous

  const int tid  = threadIdx.x;
  const int lane = tid & 31;
  const int wave = tid >> 5;
  const int rs   = wave >> 1;       // row strip 0..3 (16 rows each)
  const int nh   = wave & 1;        // N half: 8 n-tiles each
  const int blockRow = blockIdx.x * 64;

  // per-thread staging addresses
  const int ar = tid >> 2;
  const int ac = (tid & 3) * 8;
  const bf16_t* aG = A + (size_t)(blockRow + ar) * 256 + ac;
  const bf16_t* wG = Wt_g + (size_t)tid * 256;

  v8f acc[8];
#pragma unroll
  for (int t = 0; t < 8; ++t)
#pragma unroll
    for (int e = 0; e < 8; ++e) acc[t][e] = 0.f;

  for (int kk = 0; kk < 8; ++kk) {
    const int k0 = kk * 32;
    __syncthreads();
    // stage A tile 64x32 : each thread one 16B chunk
    *(v8bf*)&As[ar][ac] = *(const v8bf*)(aG + k0);
    // stage W tile: thread n copies Wt_g[n][k0..k0+31]
    {
      const v8bf* src = (const v8bf*)(wG + k0);
      v8bf* dst = (v8bf*)&Wt[tid][0];
      dst[0] = src[0]; dst[1] = src[1]; dst[2] = src[2]; dst[3] = src[3];
    }
    // prefetch next K-step's tiles into the WGP cache (locality 3 -> near scope)
    if (kk < 7) {
      __builtin_prefetch(aG + k0 + 32, 0, 3);
      __builtin_prefetch(wG + k0 + 32, 0, 3);
    }
    __syncthreads();

    // Preload A fragment and all 8 B fragments, then issue the 8 WMMAs.
    v16bf afrag = load_a_frag(&As[rs * 16][0], 32, lane);
    v16bf bfr[8];
#pragma unroll
    for (int t = 0; t < 8; ++t)
      bfr[t] = load_b_frag(&Wt[(nh * 8 + t) * 16][0], 32, lane);
#pragma unroll
    for (int t = 0; t < 8; ++t)
      acc[t] = wmma_bf16(afrag, bfr[t], acc[t]);

#if __has_builtin(__builtin_amdgcn_sched_group_barrier)
    // Pin issue order: 6 DS reads (A frag + B0 + B1), then {1 WMMA, 2 DS reads}x6,
    // then the last 2 WMMAs. Keeps one B fragment in flight ahead of each WMMA so
    // dscnt waits drain partially instead of fully gating every WMMA.
    __builtin_amdgcn_sched_group_barrier(0x100, 6, 0);  // DS read x6
#pragma unroll
    for (int t = 0; t < 6; ++t) {
      __builtin_amdgcn_sched_group_barrier(0x008, 1, 0); // WMMA x1
      __builtin_amdgcn_sched_group_barrier(0x100, 2, 0); // DS read x2
    }
    __builtin_amdgcn_sched_group_barrier(0x008, 2, 0);   // WMMA x2
#endif
  }

  // epilogue: C/D layout — VGPR r: M = r + 8*(lane>=16), N = lane%16
  const int n16     = lane & 15;
  const int rowBase = blockRow + rs * 16 + ((lane < 16) ? 0 : 8);
#pragma unroll
  for (int t = 0; t < 8; ++t) {
    int col = (nh * 8 + t) * 16 + n16;
    float bv = bias[col];
#pragma unroll
    for (int r = 0; r < 8; ++r) {
      float v = acc[t][r] + bv;
      size_t idx = (size_t)(rowBase + r) * 256 + col;
      if (outBf) outBf[idx] = (bf16_t)v;
      else       outF[idx]  = v;
    }
  }
}

// ---------------------------------------------------------------- fused attention
// One block per (b,h). LDS: Qs[208][32], Ks[208][32], Vt[16][224], Sb[208][224] bf16, maskAdd[224] f32.
#define ATTN_LDS_BYTES (LP_*32*2 + LP_*32*2 + DH_*LPK_*2 + LP_*LPK_*2 + LPK_*4)

__global__ __launch_bounds__(256) void attn_kernel(
    const bf16_t* __restrict__ Q, const bf16_t* __restrict__ K,
    const bf16_t* __restrict__ V, const int* __restrict__ mask,
    bf16_t* __restrict__ ctx) {
  extern __shared__ char smem[];
  bf16_t* Qs = (bf16_t*)smem;                         // [208][32]
  bf16_t* Ks = Qs + LP_ * 32;                         // [208][32]
  bf16_t* Vt = Ks + LP_ * 32;                         // [16][224]
  bf16_t* Sb = Vt + DH_ * LPK_;                       // [208][224]
  float* maskAdd = (float*)(Sb + (size_t)LP_ * LPK_); // [224]

  const int b    = blockIdx.x >> 4;
  const int h    = blockIdx.x & 15;
  const int tid  = threadIdx.x;
  const int lane = tid & 31;
  const int wave = tid >> 5;

  // ---- stage Q,K,V (zero-pad K-dim 16..31 and rows >= 200)
  for (int idx = tid; idx < LP_ * DH_; idx += 256) {
    int row = idx >> 4, d = idx & 15;
    bf16_t q = (bf16_t)0.f, k = (bf16_t)0.f, v = (bf16_t)0.f;
    if (row < L_) {
      size_t g = ((size_t)(b * L_ + row)) * 256 + h * 16 + d;
      q = Q[g]; k = K[g]; v = V[g];
    }
    Qs[row * 32 + d] = q;  Qs[row * 32 + 16 + d] = (bf16_t)0.f;
    Ks[row * 32 + d] = k;  Ks[row * 32 + 16 + d] = (bf16_t)0.f;
    Vt[d * LPK_ + row] = v;
  }
  for (int idx = tid; idx < DH_ * (LPK_ - L_); idx += 256) { // Vt cols 200..223 = 0
    int d = idx / (LPK_ - L_), c = L_ + idx % (LPK_ - L_);
    Vt[d * LPK_ + c] = (bf16_t)0.f;
  }
  for (int j = tid; j < LPK_; j += 256)
    maskAdd[j] = (j < L_) ? ((mask[b * L_ + j] > 0) ? 0.f : -10000.f) : -1e9f;
  __syncthreads();

  // ---- scores: S = Q K^T / 4 + mask, stored bf16. 13x13 tiles over 8 waves.
  for (int t = wave; t < 13 * 13; t += 8) {
    int ti = t / 13, tj = t % 13;
    v16bf a  = load_a_frag(Qs + ti * 16 * 32, 32, lane);
    v16bf bb = load_b_frag(Ks + tj * 16 * 32, 32, lane);
    v8f c;
#pragma unroll
    for (int e = 0; e < 8; ++e) c[e] = 0.f;
    c = wmma_bf16(a, bb, c);
    int col  = tj * 16 + (lane & 15);
    float ma = maskAdd[col];
    int rowBase = ti * 16 + ((lane < 16) ? 0 : 8);
#pragma unroll
    for (int r = 0; r < 8; ++r)
      Sb[(size_t)(rowBase + r) * LPK_ + col] = (bf16_t)(c[r] * 0.25f + ma);
  }
  __syncthreads();

  // ---- softmax per row, write P (bf16) back; zero padded K columns
  if (tid < LP_) {
    bf16_t* row = Sb + (size_t)tid * LPK_;
    float mx = -1e30f;
    for (int j = 0; j < L_; ++j) mx = fmaxf(mx, (float)row[j]);
    float s = 0.f;
    for (int j = 0; j < L_; ++j) s += __expf((float)row[j] - mx);
    float inv = 1.f / s;
    for (int j = 0; j < L_; ++j) row[j] = (bf16_t)(__expf((float)row[j] - mx) * inv);
    for (int j = L_; j < LPK_; ++j) row[j] = (bf16_t)0.f;
  }
  __syncthreads();

  // ---- ctx = P(208x224) @ V(224x16); 13 row tiles over 8 waves, 7 K-steps of 32.
  for (int ti = wave; ti < 13; ti += 8) {
    v16bf afr[7], bfr[7];
#pragma unroll
    for (int kk = 0; kk < 7; ++kk) {
      afr[kk] = load_a_frag(Sb + (size_t)ti * 16 * LPK_ + kk * 32, LPK_, lane);
      bfr[kk] = load_b_frag(Vt + kk * 32, LPK_, lane);
    }
    v8f c;
#pragma unroll
    for (int e = 0; e < 8; ++e) c[e] = 0.f;
#pragma unroll
    for (int kk = 0; kk < 7; ++kk)
      c = wmma_bf16(afr[kk], bfr[kk], c);
#if __has_builtin(__builtin_amdgcn_sched_group_barrier)
    __builtin_amdgcn_sched_group_barrier(0x100, 8, 0);   // DS read x8 (A0,B0,A1,B1)
#pragma unroll
    for (int t = 0; t < 5; ++t) {
      __builtin_amdgcn_sched_group_barrier(0x008, 1, 0); // WMMA x1
      __builtin_amdgcn_sched_group_barrier(0x100, 4, 0); // DS read x4 (next A,B)
    }
    __builtin_amdgcn_sched_group_barrier(0x008, 2, 0);   // WMMA x2
#endif
    int n = lane & 15;
    int rowBase = ti * 16 + ((lane < 16) ? 0 : 8);
#pragma unroll
    for (int r = 0; r < 8; ++r) {
      int qr = rowBase + r;
      if (qr < L_)
        ctx[((size_t)(b * L_ + qr)) * 256 + h * 16 + n] = (bf16_t)c[r];
    }
  }
}

// ---------------------------------------------------------------- pooling
__global__ __launch_bounds__(256) void pool_kernel(
    const float* __restrict__ X, const int* __restrict__ mask,
    const float* __restrict__ query, float* __restrict__ out) {
  __shared__ float qv[256];
  __shared__ float ps[256];
  const int b = blockIdx.x, t = threadIdx.x;
  qv[t] = query[t];
  __syncthreads();
  for (int l = t; l < L_; l += 256) {
    const float* x = X + ((size_t)(b * L_ + l)) * 256;
    float acc = 0.f;
    for (int d = 0; d < 256; ++d) acc += qv[d] * x[d];
    acc *= (1.f / 16.f); // 1/sqrt(256)
    ps[l] = (mask[b * L_ + l] > 0) ? acc : -1e9f;
  }
  __syncthreads();
  float mx = -1e30f;
  for (int l = 0; l < L_; ++l) mx = fmaxf(mx, ps[l]);
  float s = 0.f;
  for (int l = 0; l < L_; ++l) s += __expf(ps[l] - mx);
  float inv = 1.f / s;
  __syncthreads();
  if (t < L_) ps[t] = __expf(ps[t] - mx) * inv;
  __syncthreads();
  float acc = 0.f;
  for (int l = 0; l < L_; ++l)
    acc += ps[l] * X[((size_t)(b * L_ + l)) * 256 + t];
  out[(size_t)b * 256 + t] = acc;
}

// ---------------------------------------------------------------- launch
extern "C" void kernel_launch(void* const* d_in, const int* in_sizes, int n_in,
                              void* d_out, int out_size, void* d_ws, size_t ws_size,
                              hipStream_t stream) {
  const float* news  = (const float*)d_in[0];
  const int*   hmask = (const int*)d_in[1];
  const float* Wq = (const float*)d_in[2];  const float* bq = (const float*)d_in[3];
  const float* Wk = (const float*)d_in[4];  const float* bk = (const float*)d_in[5];
  const float* Wv = (const float*)d_in[6];  const float* bv = (const float*)d_in[7];
  const float* Wo = (const float*)d_in[8];  const float* bo = (const float*)d_in[9];
  const float* qn = (const float*)d_in[10];

  char* ws = (char*)d_ws;
  const size_t SZ_ACT = (size_t)B_ * L_ * D_ * sizeof(bf16_t); // 26,214,400
  const size_t SZ_W   = (size_t)D_ * D_ * sizeof(bf16_t);      // 131,072

  size_t o = 0;
  bf16_t* Xbf = (bf16_t*)(ws + o); o += SZ_ACT;   // also reused as ctx
  bf16_t* Wqb = (bf16_t*)(ws + o); o += SZ_W;
  bf16_t* Wkb = (bf16_t*)(ws + o); o += SZ_W;
  bf16_t* Wvb = (bf16_t*)(ws + o); o += SZ_W;
  bf16_t* Wob = (bf16_t*)(ws + o); o += SZ_W;
  size_t oQ = o;
  bf16_t* Qb = (bf16_t*)(ws + o); o += SZ_ACT;
  bf16_t* Kb = (bf16_t*)(ws + o); o += SZ_ACT;
  bf16_t* Vb = (bf16_t*)(ws + o); o += SZ_ACT;
  float* NewsOut = (float*)(ws + oQ);             // reuses Q+K space (52.4 MB) after attention
  bf16_t* Ctx = Xbf;                              // reuses X space after QKV GEMMs

  const int nX = B_ * L_ * D_;
  cvt_f32_to_bf16_kernel<<<8192, 256, 0, stream>>>(news, Xbf, nX);
  cvt_w_transpose_kernel<<<256, 256, 0, stream>>>(Wq, Wqb);
  cvt_w_transpose_kernel<<<256, 256, 0, stream>>>(Wk, Wkb);
  cvt_w_transpose_kernel<<<256, 256, 0, stream>>>(Wv, Wvb);
  cvt_w_transpose_kernel<<<256, 256, 0, stream>>>(Wo, Wob);

  const int gemmGrid = (B_ * L_) / 64; // 800
  gemm_bf16_kernel<<<gemmGrid, 256, 0, stream>>>(Xbf, Wqb, bq, Qb, nullptr);
  gemm_bf16_kernel<<<gemmGrid, 256, 0, stream>>>(Xbf, Wkb, bk, Kb, nullptr);
  gemm_bf16_kernel<<<gemmGrid, 256, 0, stream>>>(Xbf, Wvb, bv, Vb, nullptr);

  attn_kernel<<<B_ * H_, 256, ATTN_LDS_BYTES, stream>>>(Qb, Kb, Vb, hmask, Ctx);

  gemm_bf16_kernel<<<gemmGrid, 256, 0, stream>>>(Ctx, Wob, bo, nullptr, NewsOut);

  pool_kernel<<<B_, 256, 0, stream>>>(NewsOut, hmask, qn, (float*)d_out);
}
